// GNNPolicy_85375359910038
// MI455X (gfx1250) — matrix operand order repaired and checked by reading, compiled, not run on hardware
//
#include <hip/hip_runtime.h>
#include <hip/hip_bf16.h>

typedef __attribute__((ext_vector_type(16))) _Float16 v16h;
typedef __attribute__((ext_vector_type(8)))  _Float16 v8h;
typedef __attribute__((ext_vector_type(8)))  float    v8f;

#define HH 128           // hidden dim
#define NPG 4096         // nodes per graph
#define NB  16           // graphs

// ---------------------------------------------------------------------------
// Small prep kernels
// ---------------------------------------------------------------------------

// Transpose f32 weight W[K][Ncols] -> f16 Wt[Ncols][Kpad] (zero pad K..Kpad)
__global__ void transpose_pad_w(const float* __restrict__ W, _Float16* __restrict__ Wt,
                                int K, int Kpad, int Ncols) {
    int idx = blockIdx.x * blockDim.x + threadIdx.x;
    int total = Ncols * Kpad;
    if (idx >= total) return;
    int n = idx / Kpad, k = idx % Kpad;
    Wt[idx] = (_Float16)(k < K ? W[k * Ncols + n] : 0.0f);
}

// x[N][8] f32 -> xh[N][32] f16 zero-padded
__global__ void pad_x_kernel(const float* __restrict__ x, _Float16* __restrict__ xh, int total) {
    int idx = blockIdx.x * blockDim.x + threadIdx.x;
    if (idx >= total) return;
    int node = idx >> 5, k = idx & 31;
    xh[idx] = (_Float16)(k < 8 ? x[node * 8 + k] : 0.0f);
}

// Per-dst sum of edge_attr + degree (for self-loop fill_value='mean')
__global__ void loop_accum_kernel(const int* __restrict__ ei, const float* __restrict__ ea,
                                  float* __restrict__ lsum, float* __restrict__ deg, int E) {
    int e = blockIdx.x * blockDim.x + threadIdx.x;
    if (e >= E) return;
    int dst = ei[E + e];
    atomicAdd(&deg[dst], 1.0f);
    atomicAdd(&lsum[(size_t)dst * 4 + 0], ea[(size_t)e * 4 + 0]);
    atomicAdd(&lsum[(size_t)dst * 4 + 1], ea[(size_t)e * 4 + 1]);
    atomicAdd(&lsum[(size_t)dst * 4 + 2], ea[(size_t)e * 4 + 2]);
    atomicAdd(&lsum[(size_t)dst * 4 + 3], ea[(size_t)e * 4 + 3]);
}

__global__ void loop_div_kernel(float* __restrict__ lsum, const float* __restrict__ deg, int totalN4) {
    int idx = blockIdx.x * blockDim.x + threadIdx.x;
    if (idx >= totalN4) return;
    float d = deg[idx >> 2];
    d = d > 1.0f ? d : 1.0f;
    lsum[idx] = lsum[idx] / d;
}

// ---------------------------------------------------------------------------
// Dual WMMA GEMM: out0 = A@W0, out1 = A@W1, A is [M][Ktot] f16 row-major,
// Wt are [128 cols][Ktot] f16 (pre-transposed). Block = 8 waves, wave w owns
// 16-col tile w; grid.x = M/16. Uses v_wmma_f32_16x16x32_f16, f32 accumulate.
// ---------------------------------------------------------------------------
__global__ void __launch_bounds__(256)
wmma_dual_gemm(const _Float16* __restrict__ Ah,
               const _Float16* __restrict__ Wt0,
               const _Float16* __restrict__ Wt1,
               float* __restrict__ out0, float* __restrict__ out1, int Ktot) {
    const int wave  = threadIdx.x >> 5;
    const int lane  = threadIdx.x & 31;
    const int laneHi = lane >> 4;          // 0: lanes 0-15, 1: lanes 16-31
    const int lml   = lane & 15;
    const int mbase = blockIdx.x * 16;
    const int nbase = wave * 16;

    v8f accL = {}; v8f accR = {};
    const _Float16* Arow  = Ah  + (size_t)(mbase + lml) * Ktot;
    const _Float16* BrowL = Wt0 + (size_t)(nbase + lml) * Ktot;
    const _Float16* BrowR = Wt1 + (size_t)(nbase + lml) * Ktot;

    for (int kt = 0; kt < Ktot; kt += 32) {
        // A 16x32 fragment: lanes0-15 rows M, K in [kt,kt+16);
        // lanes16-31 K in [kt+16,kt+32)
        v8h alo = *(const v8h*)(Arow + kt + laneHi * 16);
        v8h ahi = *(const v8h*)(Arow + kt + laneHi * 16 + 8);
        v16h a;
        #pragma unroll
        for (int j = 0; j < 8; ++j) { a[j] = alo[j]; a[8 + j] = ahi[j]; }

        // B 32x16 fragment: lane = col (0-15 K=kt..kt+15, 16-31 K=kt+16..kt+31)
        v8h b0 = *(const v8h*)(BrowL + kt + laneHi * 16);
        v8h b1 = *(const v8h*)(BrowL + kt + laneHi * 16 + 8);
        v16h bl;
        #pragma unroll
        for (int j = 0; j < 8; ++j) { bl[j] = b0[j]; bl[8 + j] = b1[j]; }
        b0 = *(const v8h*)(BrowR + kt + laneHi * 16);
        b1 = *(const v8h*)(BrowR + kt + laneHi * 16 + 8);
        v16h br;
        #pragma unroll
        for (int j = 0; j < 8; ++j) { br[j] = b0[j]; br[8 + j] = b1[j]; }

        accL = __builtin_amdgcn_wmma_f32_16x16x32_f16(false, a, false, bl, (short)0, accL, false, false);
        accR = __builtin_amdgcn_wmma_f32_16x16x32_f16(false, a, false, br, (short)0, accR, false, false);
    }

    // C/D layout: lanes 0-15 vgpr r -> M=r; lanes 16-31 vgpr r -> M=r+8; N=lane&15
    const int col = nbase + lml;
    #pragma unroll
    for (int r = 0; r < 8; ++r) {
        int row = mbase + laneHi * 8 + r;
        out0[(size_t)row * HH + col] = accL[r];
        out1[(size_t)row * HH + col] = accR[r];
    }
}

// ---------------------------------------------------------------------------
// Edge pass 1: s = att . leaky_relu(xl[src] + xr[dst] + ea@We); atomicMax smax
// Wave-per-edge (32 lanes x 4 floats = 128 dims, coalesced 512B gathers).
// Edge ids >= Eorig are self-loops (src=dst=e-Eorig, ea=loop_attr).
// ---------------------------------------------------------------------------
__global__ void edge_score_kernel(const int* __restrict__ ei, const float* __restrict__ ea,
                                  const float* __restrict__ loop_attr,
                                  const float* __restrict__ xl, const float* __restrict__ xr,
                                  const float* __restrict__ We, const float* __restrict__ att,
                                  float* __restrict__ s_out, unsigned int* __restrict__ smax,
                                  int Eorig, int Etot) {
    int wid  = (blockIdx.x * blockDim.x + threadIdx.x) >> 5;
    int lane = threadIdx.x & 31;
    if (wid >= Etot) return;
    int src, dst; const float* e4;
    if (wid < Eorig) { src = ei[wid]; dst = ei[Eorig + wid]; e4 = ea + (size_t)wid * 4; }
    else             { src = dst = wid - Eorig;             e4 = loop_attr + (size_t)src * 4; }
    float e0 = e4[0], e1 = e4[1], e2 = e4[2], e3 = e4[3];

    int d0 = lane * 4;
    const float* xls = xl + (size_t)src * HH + d0;
    const float* xrd = xr + (size_t)dst * HH + d0;
    float p = 0.0f;
    #pragma unroll
    for (int i = 0; i < 4; ++i) {
        int d = d0 + i;
        float m = xls[i] + xrd[i] + e0 * We[d] + e1 * We[HH + d] + e2 * We[2 * HH + d] + e3 * We[3 * HH + d];
        m = m > 0.0f ? m : 0.2f * m;
        p += m * att[d];
    }
    #pragma unroll
    for (int off = 16; off > 0; off >>= 1) p += __shfl_xor(p, off, 32);
    if (lane == 0) {
        s_out[wid] = p;
        unsigned u = __float_as_uint(p);
        u = (u & 0x80000000u) ? ~u : (u | 0x80000000u);   // order-preserving encode
        atomicMax(&smax[dst], u);
    }
}

// Edge pass 2: w = exp(s - smax[dst]); denom[dst]+=w; num[dst]+=w*xl[src]
__global__ void edge_accum_kernel(const int* __restrict__ ei, const float* __restrict__ xl,
                                  const float* __restrict__ s_in, const unsigned int* __restrict__ smax,
                                  float* __restrict__ denom, float* __restrict__ num,
                                  int Eorig, int Etot) {
    int wid  = (blockIdx.x * blockDim.x + threadIdx.x) >> 5;
    int lane = threadIdx.x & 31;
    if (wid >= Etot) return;
    int src, dst;
    if (wid < Eorig) { src = ei[wid]; dst = ei[Eorig + wid]; }
    else             { src = dst = wid - Eorig; }
    unsigned u = smax[dst];
    float mx = (u & 0x80000000u) ? __uint_as_float(u & 0x7FFFFFFFu) : __uint_as_float(~u);
    float w = __expf(s_in[wid] - mx);
    if (lane == 0) atomicAdd(&denom[dst], w);
    int d0 = lane * 4;
    const float* v = xl + (size_t)src * HH + d0;
    float* np = num + (size_t)dst * HH + d0;
    atomicAdd(np + 0, w * v[0]);
    atomicAdd(np + 1, w * v[1]);
    atomicAdd(np + 2, w * v[2]);
    atomicAdd(np + 3, w * v[3]);
}

// Layer-1 finalize: h = relu(num/denom + b); emit f16 for layer-2 WMMA
__global__ void finalize1_kernel(const float* __restrict__ num, const float* __restrict__ denom,
                                 const float* __restrict__ b, _Float16* __restrict__ h_half, int total) {
    int idx = blockIdx.x * blockDim.x + threadIdx.x;
    if (idx >= total) return;
    float h = num[idx] / denom[idx >> 7] + b[idx & 127];
    h = h > 0.0f ? h : 0.0f;
    h_half[idx] = (_Float16)h;
}

// Layer-2 finalize in place: h2 = num/denom + b
__global__ void finalize2_kernel(float* __restrict__ num, const float* __restrict__ denom,
                                 const float* __restrict__ b, int total) {
    int idx = blockIdx.x * blockDim.x + threadIdx.x;
    if (idx >= total) return;
    num[idx] = num[idx] / denom[idx >> 7] + b[idx & 127];
}

// ---------------------------------------------------------------------------
// Pool + heads: one block per graph.
// ---------------------------------------------------------------------------
__global__ void __launch_bounds__(256)
head_kernel(const float* __restrict__ h2, const float* __restrict__ gs,
            const int* __restrict__ pipe,
            const float* __restrict__ Wc,  const float* __restrict__ bc,
            const float* __restrict__ Wa1, const float* __restrict__ ba1,
            const float* __restrict__ Wa2, const float* __restrict__ ba2,
            const float* __restrict__ Wv1, const float* __restrict__ bv1,
            const float* __restrict__ Wv2, const float* __restrict__ bv2,
            float* __restrict__ out) {
    __shared__ float comb[400];
    __shared__ float feat[256];
    __shared__ float hid[256];   // [0,128): a1, [128,256): v1
    const int b = blockIdx.x, t = threadIdx.x;
    const float* hb = h2 + (size_t)b * NPG * HH;

    if (t < 128) {
        float acc = 0.0f;
        for (int n = 0; n < NPG; ++n) acc += hb[(size_t)n * HH + t];
        comb[t] = acc * (1.0f / (float)NPG);
        int i1 = pipe[b * 2 + 1];
        comb[256 + t] = hb[(size_t)i1 * HH + t];
    } else if (t < 256) {
        int d = t - 128;
        int i0 = pipe[b * 2 + 0];
        comb[128 + d] = hb[(size_t)i0 * HH + d];
        if (d < 16) comb[384 + d] = gs[b * 16 + d];
    }
    __syncthreads();
    {
        float acc = bc[t];
        for (int k = 0; k < 400; ++k) acc += comb[k] * Wc[k * 256 + t];
        feat[t] = acc > 0.0f ? acc : 0.0f;
    }
    __syncthreads();
    if (t < 128) {
        float acc = ba1[t];
        for (int k = 0; k < 256; ++k) acc += feat[k] * Wa1[k * 128 + t];
        hid[t] = acc > 0.0f ? acc : 0.0f;
    } else {
        int d = t - 128;
        float acc = bv1[d];
        for (int k = 0; k < 256; ++k) acc += feat[k] * Wv1[k * 128 + d];
        hid[128 + d] = acc > 0.0f ? acc : 0.0f;
    }
    __syncthreads();
    if (t < 10) {
        float acc = ba2[t];
        for (int k = 0; k < 128; ++k) acc += hid[k] * Wa2[k * 10 + t];
        out[b * 11 + t] = acc;
    } else if (t == 10) {
        float acc = bv2[0];
        for (int k = 0; k < 128; ++k) acc += hid[128 + k] * Wv2[k];
        out[b * 11 + 10] = acc;
    }
}

// ---------------------------------------------------------------------------
extern "C" void kernel_launch(void* const* d_in, const int* in_sizes, int n_in,
                              void* d_out, int out_size, void* d_ws, size_t ws_size,
                              hipStream_t stream) {
    const float* x    = (const float*)d_in[0];
    const int*   ei   = (const int*)  d_in[1];
    const float* ea   = (const float*)d_in[2];
    const float* gs   = (const float*)d_in[3];
    const int*   pipe = (const int*)  d_in[4];
    const float* W_l1 = (const float*)d_in[5];
    const float* W_r1 = (const float*)d_in[6];
    const float* W_e1 = (const float*)d_in[7];
    const float* att1 = (const float*)d_in[8];
    const float* b1   = (const float*)d_in[9];
    const float* W_l2 = (const float*)d_in[10];
    const float* W_r2 = (const float*)d_in[11];
    const float* W_e2 = (const float*)d_in[12];
    const float* att2 = (const float*)d_in[13];
    const float* b2   = (const float*)d_in[14];
    const float* W_c  = (const float*)d_in[15];
    const float* b_c  = (const float*)d_in[16];
    const float* W_a1 = (const float*)d_in[17];
    const float* b_a1 = (const float*)d_in[18];
    const float* W_a2 = (const float*)d_in[19];
    const float* b_a2 = (const float*)d_in[20];
    const float* W_v1 = (const float*)d_in[21];
    const float* b_v1 = (const float*)d_in[22];
    const float* W_v2 = (const float*)d_in[23];
    const float* b_v2 = (const float*)d_in[24];
    float* out = (float*)d_out;

    const int N    = in_sizes[0] / 8;       // 65536
    const int E    = in_sizes[1] / 2;       // 524288
    const int Etot = E + N;                 // + self loops

    // ---- workspace carve-out (256B aligned) ----
    char* base = (char*)d_ws;
    size_t o = 0;
    auto carve = [&](size_t bytes) { char* p = base + o; o = (o + bytes + 255) & ~(size_t)255; return p; };
    float*      xl      = (float*)     carve((size_t)N * HH * 4);
    float*      xr      = (float*)     carve((size_t)N * HH * 4);
    float*      num     = (float*)     carve((size_t)N * HH * 4);   // becomes h2 in-place
    _Float16*   h_half  = (_Float16*)  carve((size_t)N * HH * 2);
    _Float16*   x_half  = (_Float16*)  carve((size_t)N * 32 * 2);
    float*      sbuf    = (float*)     carve((size_t)Etot * 4);
    float*      lattr   = (float*)     carve((size_t)N * 4 * 4);
    float*      deg     = (float*)     carve((size_t)N * 4);
    unsigned*   smax    = (unsigned*)  carve((size_t)N * 4);
    float*      denom   = (float*)     carve((size_t)N * 4);
    _Float16*   w1lt    = (_Float16*)  carve(128 * 32 * 2);
    _Float16*   w1rt    = (_Float16*)  carve(128 * 32 * 2);
    _Float16*   w2lt    = (_Float16*)  carve(128 * 128 * 2);
    _Float16*   w2rt    = (_Float16*)  carve(128 * 128 * 2);
    (void)ws_size;

    // ---- zero-init accumulators (0x00 == 0.0f; 0u is below all ordered-float codes) ----
    hipMemsetAsync(deg,   0, (size_t)N * 4,        stream);
    hipMemsetAsync(lattr, 0, (size_t)N * 4 * 4,    stream);
    hipMemsetAsync(smax,  0, (size_t)N * 4,        stream);
    hipMemsetAsync(denom, 0, (size_t)N * 4,        stream);
    hipMemsetAsync(num,   0, (size_t)N * HH * 4,   stream);

    // ---- prep: weights -> f16 transposed(+padded), x -> f16 padded, loop attrs ----
    transpose_pad_w<<<(128 * 32  + 255) / 256, 256, 0, stream>>>(W_l1, w1lt, 8,   32,  128);
    transpose_pad_w<<<(128 * 32  + 255) / 256, 256, 0, stream>>>(W_r1, w1rt, 8,   32,  128);
    transpose_pad_w<<<(128 * 128 + 255) / 256, 256, 0, stream>>>(W_l2, w2lt, 128, 128, 128);
    transpose_pad_w<<<(128 * 128 + 255) / 256, 256, 0, stream>>>(W_r2, w2rt, 128, 128, 128);
    pad_x_kernel<<<((size_t)N * 32 + 255) / 256, 256, 0, stream>>>(x, x_half, N * 32);
    loop_accum_kernel<<<(E + 255) / 256, 256, 0, stream>>>(ei, ea, lattr, deg, E);
    loop_div_kernel<<<(N * 4 + 255) / 256, 256, 0, stream>>>(lattr, deg, N * 4);

    const int edgeBlocks = Etot / 8;              // 8 waves (edges) per 256-thread block
    const int nodeElems  = N * HH;

    // ---- Layer 1 ----
    wmma_dual_gemm<<<N / 16, 256, 0, stream>>>(x_half, w1lt, w1rt, xl, xr, 32);
    edge_score_kernel<<<edgeBlocks, 256, 0, stream>>>(ei, ea, lattr, xl, xr, W_e1, att1, sbuf, smax, E, Etot);
    edge_accum_kernel<<<edgeBlocks, 256, 0, stream>>>(ei, xl, sbuf, smax, denom, num, E, Etot);
    finalize1_kernel<<<(nodeElems + 255) / 256, 256, 0, stream>>>(num, denom, b1, h_half, nodeElems);

    // reset per-dst accumulators for layer 2
    hipMemsetAsync(smax,  0, (size_t)N * 4,      stream);
    hipMemsetAsync(denom, 0, (size_t)N * 4,      stream);
    hipMemsetAsync(num,   0, (size_t)N * HH * 4, stream);

    // ---- Layer 2 ----
    wmma_dual_gemm<<<N / 16, 256, 0, stream>>>(h_half, w2lt, w2rt, xl, xr, 128);
    edge_score_kernel<<<edgeBlocks, 256, 0, stream>>>(ei, ea, lattr, xl, xr, W_e2, att2, sbuf, smax, E, Etot);
    edge_accum_kernel<<<edgeBlocks, 256, 0, stream>>>(ei, xl, sbuf, smax, denom, num, E, Etot);
    finalize2_kernel<<<(nodeElems + 255) / 256, 256, 0, stream>>>(num, denom, b2, nodeElems);

    // ---- Pool + heads ----
    head_kernel<<<NB, 256, 0, stream>>>(num, gs, pipe, W_c, b_c, W_a1, b_a1, W_a2, b_a2,
                                        W_v1, b_v1, W_v2, b_v2, out);
}
